// SelfAttentionResBlock_9045201126024
// MI455X (gfx1250) — matrix operand order repaired and checked by reading, not compiled, not run on hardware
//
#include <hip/hip_runtime.h>
#include <hip/hip_bf16.h>

// ---------------------------------------------------------------------------
// CDNA5 (gfx1250) wave32 WMMA types
// ---------------------------------------------------------------------------
typedef __attribute__((ext_vector_type(16))) __bf16 v16bf;
typedef __attribute__((ext_vector_type(8)))  __bf16 v8bf;
typedef __attribute__((ext_vector_type(8)))  float  v8f;

__device__ __forceinline__ v8f wmma_bf16(v16bf a, v16bf b, v8f c) {
  // D(16x16,f32) = A(16x32,bf16) x B(32x16,bf16) + C
  return __builtin_amdgcn_wmma_f32_16x16x32_bf16(
      /*neg_a=*/false, a, /*neg_b=*/false, b,
      /*c_mod=*/(short)0, c, /*reuse_a=*/false, /*reuse_b=*/false);
}

__device__ __forceinline__ unsigned short f2bf(float f) {
  unsigned u = __float_as_uint(f);
  u += 0x7fffu + ((u >> 16) & 1u);   // round-to-nearest-even
  return (unsigned short)(u >> 16);
}

__device__ __forceinline__ float silu(float v) {
  return v / (1.f + __expf(-v));
}

// Shapes (fixed by the reference)
#define BB   16
#define CC   256
#define HH   32
#define WW   32
#define HWQ  1024          // H*W == T
#define PW   34            // padded width/height
#define PHW  (PW * PW)     // 1156 padded positions
#define EE   1024
#define NHEAD 4
#define HD   64

// zero a bf16(ushort) buffer
__global__ void k_zero(unsigned short* __restrict__ p, size_t n) {
  size_t i = (size_t)blockIdx.x * 256 + threadIdx.x;
  if (i < n) p[i] = 0;
}

// ---------------------------------------------------------------------------
// GN1 + SiLU  (x[B,C,HW] -> a1 bf16 padded [B,PHW,C]), xb = bf16(x) padded
// ---------------------------------------------------------------------------
__global__ void k_gn1(const float* __restrict__ x, const float* __restrict__ g,
                      const float* __restrict__ bt, unsigned short* __restrict__ a1,
                      unsigned short* __restrict__ xb) {
  const int GS = 8 * HWQ;
  int b = blockIdx.x >> 5, grp = blockIdx.x & 31;
  int tid = threadIdx.x;
  const float* xp = x + ((size_t)b * CC + grp * 8) * HWQ;
  __shared__ float s1[256], s2[256];
  float sum = 0.f, sq = 0.f;
  for (int e = tid; e < GS; e += 256) { float v = xp[e]; sum += v; sq += v * v; }
  s1[tid] = sum; s2[tid] = sq; __syncthreads();
  for (int st = 128; st > 0; st >>= 1) {
    if (tid < st) { s1[tid] += s1[tid + st]; s2[tid] += s2[tid + st]; }
    __syncthreads();
  }
  float mean = s1[0] * (1.f / GS);
  float var  = s2[0] * (1.f / GS) - mean * mean;
  float rinv = rsqrtf(var + 1e-5f);
  for (int e = tid; e < GS; e += 256) {
    int cl = e >> 10, p = e & 1023, c = grp * 8 + cl;
    int py = p >> 5, px = p & 31;
    float raw = xp[e];
    float v = (raw - mean) * rinv * g[c] + bt[c];
    v = silu(v);
    size_t oi = ((size_t)b * PHW + (py + 1) * PW + (px + 1)) * CC + c;
    a1[oi] = f2bf(v);
    xb[oi] = f2bf(raw);
  }
}

// ---------------------------------------------------------------------------
// Pack conv weights OIHW fp32 -> WMMA A-fragment order bf16:
//   wp[(((mt*3+ky)*3+kx)*8 + kc)*32 + lane][16]
// lane<16: elems 0..15 -> K=0..15   (M = lane%16)
// lane>=16: elems 0..15 -> K=16..31
// ---------------------------------------------------------------------------
__global__ void k_pack_w(const float* __restrict__ w, unsigned short* __restrict__ wp) {
  size_t t = (size_t)blockIdx.x * 256 + threadIdx.x;
  const size_t total = (size_t)CC * CC * 9;
  if (t >= total) return;
  int s    = (int)(t & 15);
  int lane = (int)((t >> 4) & 31);
  size_t rest = t >> 9;
  int kc = (int)(rest & 7);  rest >>= 3;          // 8 K-chunks of 32
  int kx = (int)(rest % 3);  rest /= 3;
  int ky = (int)(rest % 3);  rest /= 3;
  int mt = (int)rest;
  int m    = lane & 15;
  int half = (lane < 16) ? 0 : 8;
  int kk   = half * 2 + s;
  int o  = mt * 16 + m;
  int ci = kc * 32 + kk;
  wp[t] = f2bf(w[(((size_t)o * CC + ci) * 3 + ky) * 3 + kx]);
}

// ---------------------------------------------------------------------------
// Implicit-GEMM 3x3 conv (WMMA). in bf16 zero-padded [B,PHW,Cin], wp packed,
// out f32 [B,Cout,HW]. One wave per 16(cout) x 16(pos) tile; branch-free
// inner loop: 2 loads + 1 WMMA, dual accumulators for ILP. 72 WMMAs/tile.
// ---------------------------------------------------------------------------
__global__ void __launch_bounds__(32)
k_conv(const unsigned short* __restrict__ in, const unsigned short* __restrict__ wp,
       const float* __restrict__ bias, float* __restrict__ out, int accum) {
  int lane = threadIdx.x;
  int nt = blockIdx.x & 63;
  int b  = blockIdx.x >> 6;
  int mt = blockIdx.y;
  int n  = lane & 15;
  int pos = nt * 16 + n;
  int py = pos >> 5, px = pos & 31;
  int bsel = (lane < 16) ? 0 : 16;
  // (py,px) halo origin: padded index (py+ky, px+kx) for ky,kx in 0..2
  const unsigned short* bbase0 = in + ((size_t)b * PHW + py * PW + px) * CC + bsel;
  v8f acc0 = {}, acc1 = {};
#pragma unroll
  for (int ky = 0; ky < 3; ++ky) {
#pragma unroll
    for (int kx = 0; kx < 3; ++kx) {
      const unsigned short* bbase = bbase0 + (size_t)(ky * PW + kx) * CC;
      const unsigned short* abase = wp + ((size_t)((mt * 3 + ky) * 3 + kx)) * 8 * 32 * 16
                                       + (size_t)lane * 16;
      __builtin_prefetch(bbase, 0, 1);   // global_prefetch_b8
#pragma unroll
      for (int kc = 0; kc < 8; kc += 2) {
        v16bf bf0 = *(const v16bf*)(bbase + kc * 32);
        v16bf af0 = *(const v16bf*)(abase + (size_t)kc * 32 * 16);
        v16bf bf1 = *(const v16bf*)(bbase + (kc + 1) * 32);
        v16bf af1 = *(const v16bf*)(abase + (size_t)(kc + 1) * 32 * 16);
        acc0 = wmma_bf16(af0, bf0, acc0);
        acc1 = wmma_bf16(af1, bf1, acc1);
      }
    }
  }
  int h8 = (lane < 16) ? 0 : 8;
  float* op = out + ((size_t)b * CC + mt * 16) * HWQ;
#pragma unroll
  for (int r = 0; r < 8; ++r) {
    int m = r + h8;
    float v = acc0[r] + acc1[r] + bias[mt * 16 + m];
    size_t idx = (size_t)m * HWQ + pos;
    if (accum) op[idx] += v; else op[idx] = v;
  }
}

// ---------------------------------------------------------------------------
// emb MLP: out[b, j] = b_emb[j] + sum_e silu(emb[b,e]) * W_emb[j,e]   (j<512)
// ---------------------------------------------------------------------------
__global__ void k_emb(const float* __restrict__ emb, const float* __restrict__ W,
                      const float* __restrict__ bb, float* __restrict__ out) {
  int j = blockIdx.x * blockDim.x + threadIdx.x;
  if (j >= BB * 512) return;
  int b = j >> 9, c = j & 511;
  const float* e = emb + (size_t)b * EE;
  const float* w = W + (size_t)c * EE;
  float s = bb[c];
  for (int k = 0; k < EE; ++k) s += silu(e[k]) * w[k];
  out[j] = s;
}

// ---------------------------------------------------------------------------
// GN2 + scale-shift + SiLU: mid f32 [B,C,HW] -> a2 bf16 padded [B,PHW,C]
// ---------------------------------------------------------------------------
__global__ void k_gn2ss(const float* __restrict__ mid, const float* __restrict__ g,
                        const float* __restrict__ bt, const float* __restrict__ embo,
                        unsigned short* __restrict__ a2) {
  const int GS = 8 * HWQ;
  int b = blockIdx.x >> 5, grp = blockIdx.x & 31;
  int tid = threadIdx.x;
  const float* xp = mid + ((size_t)b * CC + grp * 8) * HWQ;
  __shared__ float s1[256], s2[256];
  float sum = 0.f, sq = 0.f;
  for (int e = tid; e < GS; e += 256) { float v = xp[e]; sum += v; sq += v * v; }
  s1[tid] = sum; s2[tid] = sq; __syncthreads();
  for (int st = 128; st > 0; st >>= 1) {
    if (tid < st) { s1[tid] += s1[tid + st]; s2[tid] += s2[tid + st]; }
    __syncthreads();
  }
  float mean = s1[0] * (1.f / GS);
  float rinv = rsqrtf(s2[0] * (1.f / GS) - mean * mean + 1e-5f);
  for (int e = tid; e < GS; e += 256) {
    int cl = e >> 10, p = e & 1023, c = grp * 8 + cl;
    int py = p >> 5, px = p & 31;
    float v = (xp[e] - mean) * rinv * g[c] + bt[c];
    float sc = embo[b * 512 + c], sh = embo[b * 512 + 256 + c];
    v = silu(v * (1.f + sc) + sh);
    a2[((size_t)b * PHW + (py + 1) * PW + (px + 1)) * CC + c] = f2bf(v);
  }
}

// ---------------------------------------------------------------------------
// GN attn: h f32 [B,C,T] -> xn f32 [B,C,T] + xnb bf16 [B,T,C]
// ---------------------------------------------------------------------------
__global__ void k_gna(const float* __restrict__ h, const float* __restrict__ g,
                      const float* __restrict__ bt, float* __restrict__ xn,
                      unsigned short* __restrict__ xnb) {
  const int GS = 8 * HWQ;
  int b = blockIdx.x >> 5, grp = blockIdx.x & 31;
  int tid = threadIdx.x;
  const float* xp = h + ((size_t)b * CC + grp * 8) * HWQ;
  __shared__ float s1[256], s2[256];
  float sum = 0.f, sq = 0.f;
  for (int e = tid; e < GS; e += 256) { float v = xp[e]; sum += v; sq += v * v; }
  s1[tid] = sum; s2[tid] = sq; __syncthreads();
  for (int st = 128; st > 0; st >>= 1) {
    if (tid < st) { s1[tid] += s1[tid + st]; s2[tid] += s2[tid + st]; }
    __syncthreads();
  }
  float mean = s1[0] * (1.f / GS);
  float rinv = rsqrtf(s2[0] * (1.f / GS) - mean * mean + 1e-5f);
  for (int e = tid; e < GS; e += 256) {
    int cl = e >> 10, p = e & 1023, c = grp * 8 + cl;
    float v = (xp[e] - mean) * rinv * g[c] + bt[c];
    xn[((size_t)b * CC + c) * HWQ + p] = v;
    xnb[((size_t)(b * HWQ) + p) * CC + c] = f2bf(v);
  }
}

// fp32 -> bf16 convert
__global__ void k_cvt(const float* __restrict__ a, unsigned short* __restrict__ o, int n) {
  int i = blockIdx.x * blockDim.x + threadIdx.x;
  if (i < n) o[i] = f2bf(a[i]);
}

// ---------------------------------------------------------------------------
// Generic row-major WMMA GEMM: D[b,t,j] = bias[j] + A[b,t,:] . Wb[j,:]
// A bf16 [B,T,K], Wb bf16 [N,K].  outMode 0: bf16 [B,T,N]; 1: f32 [B,N,T]
// ---------------------------------------------------------------------------
__global__ void __launch_bounds__(32)
k_gemm(const unsigned short* __restrict__ A, const unsigned short* __restrict__ Wb,
       const float* __restrict__ bias, void* __restrict__ D, int K, int N, int outMode) {
  int lane = threadIdx.x;
  int tt = blockIdx.x & 63;
  int b  = blockIdx.x >> 6;
  int j0 = blockIdx.y * 16;
  int t0 = tt * 16;
  int n  = lane & 15;
  int half = (lane < 16) ? 0 : 8;
  int bsel = (lane < 16) ? 0 : 16;
  const unsigned short* arow = A  + ((size_t)(b * HWQ) + t0 + n) * K;
  const unsigned short* brow = Wb + ((size_t)(j0 + n)) * K;
  v8f acc = {};
  for (int k0 = 0; k0 < K; k0 += 32) {
    union { v16bf v; v8bf h[2]; } ua;
    ua.h[0] = *(const v8bf*)(arow + k0 + bsel);
    ua.h[1] = *(const v8bf*)(arow + k0 + bsel + 8);
    v16bf bf = *(const v16bf*)(brow + k0 + bsel);
    acc = wmma_bf16(ua.v, bf, acc);
  }
  int h8 = (lane < 16) ? 0 : 8;
#pragma unroll
  for (int r = 0; r < 8; ++r) {
    int t = t0 + r + h8;
    int j = j0 + n;
    float v = acc[r] + bias[j];
    if (outMode == 0)
      ((unsigned short*)D)[((size_t)(b * HWQ) + t) * N + j] = f2bf(v);
    else
      ((float*)D)[((size_t)b * N + j) * HWQ + t] = v;
  }
}

// transpose V out of qkv: vT[((b*4+h)*64+d)*T + s] = qkv[(b*T+s)*768 + 512 + h*64 + d]
__global__ void k_vT(const unsigned short* __restrict__ qkv, unsigned short* __restrict__ vT) {
  int i = blockIdx.x * 256 + threadIdx.x;   // 16*4*64*1024 total
  int s = i & 1023, d = (i >> 10) & 63, h = (i >> 16) & 3, b = i >> 18;
  vT[i] = qkv[((size_t)(b * HWQ) + s) * 768 + 512 + h * HD + d];
}

// ---------------------------------------------------------------------------
// Flash attention: one wave per (b, head, 16-query tile). Streams 32 keys/step.
// S via 4 WMMAs, online softmax (shfl_xor row reductions), P re-layout via LDS,
// O += P@V via 4 full-K WMMAs against vT [d][s].
// ---------------------------------------------------------------------------
__global__ void __launch_bounds__(32)
k_attn(const unsigned short* __restrict__ qkv, const unsigned short* __restrict__ vT,
       unsigned short* __restrict__ ob) {
  int lane = threadIdx.x;
  int tt = blockIdx.x & 63;
  int hh = (blockIdx.x >> 6) & 3;
  int b  = blockIdx.x >> 8;
  int t0 = tt * 16;
  int n  = lane & 15;
  int half = (lane < 16) ? 0 : 8;
  int bsel = (lane < 16) ? 0 : 16;
  int h8 = half;
  __shared__ __align__(64) unsigned short Plds[16][32];
  const float scale = 0.125f;   // 1/sqrt(64)
  const unsigned short* qrow = qkv + ((size_t)(b * HWQ) + t0 + n) * 768 + hh * HD;
  const unsigned short* vbase = vT + (((size_t)(b * NHEAD) + hh) * HD) * HWQ;
  float mrow[8], lrow[8];
  v8f o0 = {}, o1 = {}, o2 = {}, o3 = {};
#pragma unroll
  for (int r = 0; r < 8; ++r) { mrow[r] = -1e30f; lrow[r] = 0.f; }

  for (int s0 = 0; s0 < HWQ; s0 += 32) {
    // ---- S = q . k^T for 32 keys (two 16-key sub-tiles) ----
    const unsigned short* krow0 = qkv + ((size_t)(b * HWQ) + s0 + n) * 768 + 256 + hh * HD;
    const unsigned short* krow1 = krow0 + 16 * 768;
    __builtin_prefetch(krow0 + 32 * 768, 0, 1);
    v8f sa = {}, sb = {};
#pragma unroll
    for (int k0 = 0; k0 < HD; k0 += 32) {
      union { v16bf v; v8bf h[2]; } ua;
      ua.h[0] = *(const v8bf*)(qrow + k0 + bsel);
      ua.h[1] = *(const v8bf*)(qrow + k0 + bsel + 8);
      v16bf b0 = *(const v16bf*)(krow0 + k0 + bsel);
      v16bf b1 = *(const v16bf*)(krow1 + k0 + bsel);
      sa = wmma_bf16(ua.v, b0, sa);
      sb = wmma_bf16(ua.v, b1, sb);
    }
    // ---- online softmax ----
    float corr[8];
#pragma unroll
    for (int r = 0; r < 8; ++r) {
      float mx = fmaxf(sa[r], sb[r]) * scale;
      for (int off = 1; off < 16; off <<= 1)
        mx = fmaxf(mx, __shfl_xor(mx, off, 32));
      float mn = fmaxf(mrow[r], mx);
      corr[r] = __expf(mrow[r] - mn);
      mrow[r] = mn;
      float p0 = __expf(sa[r] * scale - mn);
      float p1 = __expf(sb[r] * scale - mn);
      int row = r + h8;
      Plds[row][n]      = f2bf(p0);
      Plds[row][16 + n] = f2bf(p1);
      float rs = p0 + p1;
      for (int off = 1; off < 16; off <<= 1)
        rs += __shfl_xor(rs, off, 32);
      lrow[r] = lrow[r] * corr[r] + rs;
      o0[r] *= corr[r]; o1[r] *= corr[r]; o2[r] *= corr[r]; o3[r] *= corr[r];
    }
    asm volatile("s_wait_dscnt 0x0" ::: "memory");   // P stores visible before re-read
    // ---- re-load P as A fragment (16x32, full K) ----
    union { v16bf v; v8bf h[2]; } pa;
    int mA = lane & 15;
    pa.h[0] = *(const v8bf*)&Plds[mA][bsel];
    pa.h[1] = *(const v8bf*)&Plds[mA][bsel + 8];
    // ---- O += P @ V  (B frag: K=s contiguous in vT) ----
    int sb0 = s0 + bsel;
    v16bf vb0 = *(const v16bf*)(vbase + ((size_t)(0  + n)) * HWQ + sb0);
    v16bf vb1 = *(const v16bf*)(vbase + ((size_t)(16 + n)) * HWQ + sb0);
    v16bf vb2 = *(const v16bf*)(vbase + ((size_t)(32 + n)) * HWQ + sb0);
    v16bf vb3 = *(const v16bf*)(vbase + ((size_t)(48 + n)) * HWQ + sb0);
    o0 = wmma_bf16(pa.v, vb0, o0);
    o1 = wmma_bf16(pa.v, vb1, o1);
    o2 = wmma_bf16(pa.v, vb2, o2);
    o3 = wmma_bf16(pa.v, vb3, o3);
  }
#pragma unroll
  for (int r = 0; r < 8; ++r) {
    float li = 1.0f / lrow[r];
    int t = t0 + r + h8;
    unsigned short* p = ob + ((size_t)(b * HWQ) + t) * CC + hh * HD;
    p[0 + n]  = f2bf(o0[r] * li);
    p[16 + n] = f2bf(o1[r] * li);
    p[32 + n] = f2bf(o2[r] * li);
    p[48 + n] = f2bf(o3[r] * li);
  }
}

// ---------------------------------------------------------------------------
// Final GN (no affine): z = xn/sqrt(2) + hob (both f32 [B,C,T]) -> out
// ---------------------------------------------------------------------------
__global__ void k_gnfinal(const float* __restrict__ xn, const float* __restrict__ hob,
                          float* __restrict__ out) {
  const int GS = 8 * HWQ;
  const float RS = 0.70710678118654752f;
  int b = blockIdx.x >> 5, grp = blockIdx.x & 31;
  int tid = threadIdx.x;
  size_t base = ((size_t)b * CC + grp * 8) * HWQ;
  __shared__ float s1[256], s2[256];
  float sum = 0.f, sq = 0.f;
  for (int e = tid; e < GS; e += 256) {
    float v = RS * xn[base + e] + hob[base + e];
    sum += v; sq += v * v;
  }
  s1[tid] = sum; s2[tid] = sq; __syncthreads();
  for (int st = 128; st > 0; st >>= 1) {
    if (tid < st) { s1[tid] += s1[tid + st]; s2[tid] += s2[tid + st]; }
    __syncthreads();
  }
  float mean = s1[0] * (1.f / GS);
  float rinv = rsqrtf(s2[0] * (1.f / GS) - mean * mean + 1e-5f);
  for (int e = tid; e < GS; e += 256) {
    float v = RS * xn[base + e] + hob[base + e];
    out[base + e] = (v - mean) * rinv;
  }
}

// ---------------------------------------------------------------------------
extern "C" void kernel_launch(void* const* d_in, const int* in_sizes, int n_in,
                              void* d_out, int out_size, void* d_ws, size_t ws_size,
                              hipStream_t stream) {
  const float* x      = (const float*)d_in[0];
  const float* emb    = (const float*)d_in[1];
  const float* gn1_g  = (const float*)d_in[2];
  const float* gn1_b  = (const float*)d_in[3];
  const float* W1     = (const float*)d_in[4];
  const float* c1     = (const float*)d_in[5];
  const float* W_emb  = (const float*)d_in[6];
  const float* b_emb  = (const float*)d_in[7];
  const float* gn2_g  = (const float*)d_in[8];
  const float* gn2_b  = (const float*)d_in[9];
  const float* W2     = (const float*)d_in[10];
  const float* c2     = (const float*)d_in[11];
  const float* Wskip  = (const float*)d_in[12];
  const float* cskip  = (const float*)d_in[13];
  const float* gna_g  = (const float*)d_in[14];
  const float* gna_b  = (const float*)d_in[15];
  const float* W_in   = (const float*)d_in[16];
  const float* b_in   = (const float*)d_in[17];
  const float* W_out  = (const float*)d_in[18];
  const float* b_out  = (const float*)d_in[19];
  float* out = (float*)d_out;

  char* ws = (char*)d_ws;
  size_t off = 0;
  auto carve = [&](size_t bytes) -> void* {
    void* p = ws + off;
    off += (bytes + 255) & ~(size_t)255;
    return p;
  };
  const size_t ACT  = (size_t)BB * HWQ * CC;   // dense activations
  const size_t ACTP = (size_t)BB * PHW * CC;   // zero-padded conv inputs
  unsigned short* xb    = (unsigned short*)carve(ACTP * 2);
  unsigned short* a1    = (unsigned short*)carve(ACTP * 2);
  unsigned short* a2    = (unsigned short*)carve(ACTP * 2);
  unsigned short* wp1   = (unsigned short*)carve((size_t)CC * CC * 9 * 2);
  unsigned short* wp2   = (unsigned short*)carve((size_t)CC * CC * 9 * 2);
  unsigned short* wps   = (unsigned short*)carve((size_t)CC * CC * 9 * 2);
  float*          mid   = (float*)carve(ACT * 4);
  float*          embo  = (float*)carve((size_t)BB * 512 * 4);
  float*          hbuf  = (float*)carve(ACT * 4);
  float*          xn    = (float*)carve(ACT * 4);
  unsigned short* xnb   = (unsigned short*)carve(ACT * 2);
  unsigned short* winb  = (unsigned short*)carve((size_t)768 * 256 * 2);
  unsigned short* woutb = (unsigned short*)carve((size_t)256 * 256 * 2);
  unsigned short* qkv   = (unsigned short*)carve((size_t)BB * HWQ * 768 * 2);
  unsigned short* vTb   = (unsigned short*)carve((size_t)BB * NHEAD * HD * HWQ * 2);
  unsigned short* attnb = (unsigned short*)carve(ACT * 2);
  float*          hob   = (float*)carve(ACT * 4);

  const int wtot = CC * CC * 9;
  const int zgrid = (int)((ACTP + 255) / 256);
  // ResBlock
  k_zero<<<zgrid, 256, 0, stream>>>(xb, ACTP);
  k_zero<<<zgrid, 256, 0, stream>>>(a1, ACTP);
  k_zero<<<zgrid, 256, 0, stream>>>(a2, ACTP);
  k_gn1<<<BB * 32, 256, 0, stream>>>(x, gn1_g, gn1_b, a1, xb);
  k_pack_w<<<(wtot + 255) / 256, 256, 0, stream>>>(W1, wp1);
  k_pack_w<<<(wtot + 255) / 256, 256, 0, stream>>>(W2, wp2);
  k_pack_w<<<(wtot + 255) / 256, 256, 0, stream>>>(Wskip, wps);
  k_conv<<<dim3(BB * 64, CC / 16), 32, 0, stream>>>(a1, wp1, c1, mid, 0);
  k_emb<<<(BB * 512 + 255) / 256, 256, 0, stream>>>(emb, W_emb, b_emb, embo);
  k_gn2ss<<<BB * 32, 256, 0, stream>>>(mid, gn2_g, gn2_b, embo, a2);
  k_conv<<<dim3(BB * 64, CC / 16), 32, 0, stream>>>(xb, wps, cskip, hbuf, 0);
  k_conv<<<dim3(BB * 64, CC / 16), 32, 0, stream>>>(a2, wp2, c2, hbuf, 1);
  // SelfAttention
  k_gna<<<BB * 32, 256, 0, stream>>>(hbuf, gna_g, gna_b, xn, xnb);
  k_cvt<<<(768 * 256 + 255) / 256, 256, 0, stream>>>(W_in, winb, 768 * 256);
  k_cvt<<<(256 * 256 + 255) / 256, 256, 0, stream>>>(W_out, woutb, 256 * 256);
  k_gemm<<<dim3(BB * 64, 768 / 16), 32, 0, stream>>>(xnb, winb, b_in, qkv, 256, 768, 0);
  k_vT<<<(BB * NHEAD * HD * HWQ) / 256, 256, 0, stream>>>(qkv, vTb);
  k_attn<<<BB * NHEAD * 64, 32, 0, stream>>>(qkv, vTb, attnb);
  k_gemm<<<dim3(BB * 64, CC / 16), 32, 0, stream>>>(attnb, woutb, b_out, hob, 256, 256, 1);
  k_gnfinal<<<BB * 32, 256, 0, stream>>>(xn, hob, out);

  (void)in_sizes; (void)n_in; (void)out_size; (void)ws_size;
}
